// ManualMoELayer_7017976561990
// MI455X (gfx1250) — compile-verified, matrix operand order checked by reading, not run on hardware
//
#include <hip/hip_runtime.h>
#include <hip/hip_bf16.h>

// ---------------------------------------------------------------------------
// Problem constants (match reference)
// ---------------------------------------------------------------------------
#define BB 2
#define TT 2048
#define CC 1024
#define EE 8
#define HH 2048
#define KTOP 2
#define NTOK (BB * TT)   // 4096 tokens

#define LDS_PITCH 40     // padded row pitch (elements): 80B, 16B-aligned, bank-conflict-free

typedef __attribute__((ext_vector_type(16))) __bf16 v16bf;
typedef __attribute__((ext_vector_type(8)))  float  v8f;

// ---------------------------------------------------------------------------
// Helpers
// ---------------------------------------------------------------------------
__device__ __forceinline__ unsigned short f32_to_bf16_rne(float f) {
    unsigned int u = __float_as_uint(f);
    unsigned int lsb = (u >> 16) & 1u;
    u += 0x7FFFu + lsb;
    return (unsigned short)(u >> 16);
}

__device__ __forceinline__ float bf16_to_f32(unsigned short s) {
    return __uint_as_float(((unsigned int)s) << 16);
}

// ---------------------------------------------------------------------------
// Async global -> LDS copy (CDNA5 GLOBAL_LOAD_ASYNC_TO_LDS_*, ASYNCcnt).
// ---------------------------------------------------------------------------
typedef int v2i __attribute__((vector_size(8)));
typedef __attribute__((address_space(1))) v2i* g_v2i_p;
typedef __attribute__((address_space(3))) v2i* l_v2i_p;

#if __has_builtin(__builtin_amdgcn_global_load_async_to_lds_b64)
#define HAVE_ASYNC_LDS 1
#else
#define HAVE_ASYNC_LDS 0
#endif

__device__ __forceinline__ void g2l_b64(const void* g, void* l) {
#if HAVE_ASYNC_LDS
    __builtin_amdgcn_global_load_async_to_lds_b64((g_v2i_p)g, (l_v2i_p)l, 0, 0);
#else
    *(int2*)l = *(const int2*)g;
#endif
}

__device__ __forceinline__ void wait_async_copies() {
#if HAVE_ASYNC_LDS
#if __has_builtin(__builtin_amdgcn_s_wait_asynccnt)
    __builtin_amdgcn_s_wait_asynccnt(0);
#else
    asm volatile("s_wait_asynccnt 0x0" ::: "memory");
#endif
#endif
}

// ---------------------------------------------------------------------------
// 1) fp32 -> bf16 bulk convert (4 elements / thread)
// ---------------------------------------------------------------------------
__global__ __launch_bounds__(256) void cvt_f32_bf16_kernel(const float* __restrict__ src,
                                                           unsigned short* __restrict__ dst,
                                                           int n4) {
    int i = blockIdx.x * blockDim.x + threadIdx.x;
    if (i >= n4) return;
    float4 f = ((const float4*)src)[i];
    unsigned long long lo = (unsigned long long)(f32_to_bf16_rne(f.x) |
                           ((unsigned int)f32_to_bf16_rne(f.y) << 16));
    unsigned long long hi = (unsigned long long)(f32_to_bf16_rne(f.z) |
                           ((unsigned int)f32_to_bf16_rne(f.w) << 16));
    ((unsigned long long*)dst)[i] = lo | (hi << 32);
}

// ---------------------------------------------------------------------------
// 2) zero the per-expert counters
// ---------------------------------------------------------------------------
__global__ void zero_counts_kernel(int* __restrict__ counts) {
    if (threadIdx.x < EE) counts[threadIdx.x] = 0;
}

// ---------------------------------------------------------------------------
// 3) Router: scores = x @ Wg, top-2, softmax over the 2, slot allocation
// ---------------------------------------------------------------------------
__global__ __launch_bounds__(128) void router_kernel(const float* __restrict__ x,
                                                     const float* __restrict__ Wg,
                                                     int*   __restrict__ counts,
                                                     int*   __restrict__ tok_idx,   // [E][N]
                                                     int*   __restrict__ tok_e,     // [N][2]
                                                     float* __restrict__ tok_p,     // [N][2]
                                                     int*   __restrict__ tok_slot)  // [N][2]
{
    int n = blockIdx.x * blockDim.x + threadIdx.x;
    if (n >= NTOK) return;

    float s[EE];
#pragma unroll
    for (int e = 0; e < EE; ++e) s[e] = 0.f;

    const float* xr = x + (size_t)n * CC;
    for (int c = 0; c < CC; ++c) {
        float xv = xr[c];
        const float* wr = Wg + (size_t)c * EE;
#pragma unroll
        for (int e = 0; e < EE; ++e) s[e] += xv * wr[e];
    }

    int i0 = 0;
#pragma unroll
    for (int e = 1; e < EE; ++e) if (s[e] > s[i0]) i0 = e;
    int i1 = (i0 == 0) ? 1 : 0;
#pragma unroll
    for (int e = 0; e < EE; ++e) if (e != i0 && s[e] > s[i1]) i1 = e;

    float e1 = __expf(s[i1] - s[i0]);   // s[i0] >= s[i1]
    float denom = 1.f + e1;
    float p0 = 1.f / denom;
    float p1 = e1 / denom;

    int slot0 = atomicAdd(&counts[i0], 1);
    tok_idx[i0 * NTOK + slot0] = n;
    int slot1 = atomicAdd(&counts[i1], 1);
    tok_idx[i1 * NTOK + slot1] = n;

    tok_e[2 * n + 0] = i0;  tok_p[2 * n + 0] = p0;  tok_slot[2 * n + 0] = slot0;
    tok_e[2 * n + 1] = i1;  tok_p[2 * n + 1] = p1;  tok_slot[2 * n + 1] = slot1;
}

// ---------------------------------------------------------------------------
// 4/5) Expert GEMM, LDS-staged, 8 waves / block, 32(M) x 256(N) block tile.
//   Per 32-wide K step:
//     A tile 32x32  -> LDS (row-major, pitch 40) via async b64 copies
//     B tile 32x256 -> LDS TRANSPOSED ([n][k], pitch 40) via b128 loads +
//                      conflict-free b16 stores (lanes span 64 contiguous B)
//   Fragments: A = 2x ds_load_b128, each B tile = 2x ds_load_b128 (no packing).
//   Pitch 40 elements (80B): bank stride 20 dwords mod 64 -> 16 lanes hit 16
//   distinct banks; 80B keeps 16B alignment for b128.
//
//   WMMA 16-bit fragment layout (wave32): VGPR v, half=lane>>4:
//     koff(v) = (v<4 ? 0 : 16) + half*8 + (v&3)*2   (A: lane&15=M, B: lane&15=N)
//   => v=0..3 are 16 contiguous bytes at half*16; v=4..7 at +32B. Same holds
//   for B when stored [n][k].
// ---------------------------------------------------------------------------
template<int KTOT, int NTOT, bool GATHER, bool SILU>
__global__ __launch_bounds__(256) void ffn_kernel(
    const unsigned short* __restrict__ Asrc,   // GATHER: xb [N][C]; else hbuf [E][N][KTOT]
    const unsigned short* __restrict__ Bw,     // [E][KTOT][NTOT]
    const int* __restrict__ counts,
    const int* __restrict__ tok_idx,           // [E][N]
    unsigned short* __restrict__ outp)         // [E][N][NTOT]
{
    __shared__ __attribute__((aligned(16))) unsigned short ldsA [32  * LDS_PITCH];  //  2.5 KB
    __shared__ __attribute__((aligned(16))) unsigned short ldsBT[256 * LDS_PITCH];  // 20 KB

    const int e   = blockIdx.z;
    const int cnt = counts[e];
    const int tokBase = blockIdx.y * 32;
    if (tokBase >= cnt) return;
    const int nBlock = blockIdx.x * 256;

    const int tid   = threadIdx.x;
    const int wave  = tid >> 5;
    const int lane  = tid & 31;
    const int waveM = wave >> 2;      // 0..1 : 16-row M strip
    const int waveN = wave & 3;       // 0..3 : 64-col N strip
    const int half  = lane >> 4;
    const int l15   = lane & 15;

    // --- staging assignments -------------------------------------------------
    // A: 32 rows x 64B; thread t -> row tid>>3, 8B chunk (tid&7)
    const int aRow   = tid >> 3;
    const int aChunk = tid & 7;
    const unsigned short* aptr;
    if (GATHER) {
        int slot = tokBase + aRow;
        if (slot >= cnt) slot = cnt - 1;               // clamp: padded rows harmless
        const int token = tok_idx[e * NTOK + slot];
        aptr = Asrc + (size_t)token * KTOT;
    } else {
        aptr = Asrc + ((size_t)e * NTOK + (tokBase + aRow)) * KTOT;
    }
    unsigned short* const adst = &ldsA[aRow * LDS_PITCH + aChunk * 4];

    // B: thread t -> k-row tid&31, 32-col chunk (tid>>5); loads 4x b128,
    // scatters transposed into ldsBT[n][k].
    const int bRow   = tid & 31;     // k
    const int bChunk = tid >> 5;     // 0..7 (32 n's each)
    const unsigned short* bptr = Bw + (size_t)e * KTOT * NTOT
                                    + (size_t)bRow * NTOT + nBlock + bChunk * 32;

    v8f acc[4] = {};

    for (int k0 = 0; k0 < KTOT; k0 += 32) {
        // ---- stage A (async global -> LDS, ASYNCcnt) ----
        g2l_b64(aptr + k0 + aChunk * 4, adst);

        // ---- stage B transposed ----
        {
            const unsigned short* bsrc = bptr + (size_t)k0 * NTOT;
            union { int4 q[4]; unsigned short s[32]; } buf;
            buf.q[0] = *(const int4*)(bsrc);
            buf.q[1] = *(const int4*)(bsrc + 8);
            buf.q[2] = *(const int4*)(bsrc + 16);
            buf.q[3] = *(const int4*)(bsrc + 24);
#pragma unroll
            for (int i = 0; i < 32; ++i) {
                ldsBT[(bChunk * 32 + i) * LDS_PITCH + bRow] = buf.s[i];
            }
        }
        wait_async_copies();
        __syncthreads();

        // ---- A fragment: two 128-bit LDS reads ----
        union { unsigned int u[8]; v16bf v; } A;
        {
            const int r = waveM * 16 + l15;
            int4 a0 = *(const int4*)&ldsA[r * LDS_PITCH + half * 8];
            int4 a1 = *(const int4*)&ldsA[r * LDS_PITCH + 16 + half * 8];
            A.u[0] = a0.x; A.u[1] = a0.y; A.u[2] = a0.z; A.u[3] = a0.w;
            A.u[4] = a1.x; A.u[5] = a1.y; A.u[6] = a1.z; A.u[7] = a1.w;
        }

        // ---- 4 N tiles share the A fragment; B fragments are 2x b128 each ----
#pragma unroll
        for (int j = 0; j < 4; ++j) {
            union { unsigned int u[8]; v16bf v; } Bf;
            const int nl = waveN * 64 + j * 16 + l15;
            int4 b0 = *(const int4*)&ldsBT[nl * LDS_PITCH + half * 8];
            int4 b1 = *(const int4*)&ldsBT[nl * LDS_PITCH + 16 + half * 8];
            Bf.u[0] = b0.x; Bf.u[1] = b0.y; Bf.u[2] = b0.z; Bf.u[3] = b0.w;
            Bf.u[4] = b1.x; Bf.u[5] = b1.y; Bf.u[6] = b1.z; Bf.u[7] = b1.w;
            acc[j] = __builtin_amdgcn_wmma_f32_16x16x32_bf16(false, A.v, false, Bf.v,
                                                             (short)0, acc[j], false, false);
        }
        __syncthreads();   // before next stage overwrites LDS
    }

    // ---- epilogue: (optional) silu, bf16 store ----
    unsigned short* const obase = outp + (size_t)e * NTOK * NTOT;
#pragma unroll
    for (int j = 0; j < 4; ++j) {
        const int col = nBlock + waveN * 64 + j * 16 + l15;
#pragma unroll
        for (int r = 0; r < 8; ++r) {
            const int row = tokBase + waveM * 16 + r + 8 * half;
            float vv = acc[j][r];
            if (SILU) vv = vv / (1.f + __expf(-vv));
            obase[(size_t)row * NTOT + col] = f32_to_bf16_rne(vv);
        }
    }
}

// ---------------------------------------------------------------------------
// 6) Combine: y[n,c] = sum_k p_k * eo[e_k, slot_k, c]  (deterministic order)
// ---------------------------------------------------------------------------
__global__ __launch_bounds__(256) void combine_kernel(const unsigned short* __restrict__ eobuf,
                                                      const int*   __restrict__ tok_e,
                                                      const float* __restrict__ tok_p,
                                                      const int*   __restrict__ tok_slot,
                                                      float* __restrict__ y)
{
    int idx = blockIdx.x * blockDim.x + threadIdx.x;
    if (idx >= NTOK * CC) return;
    int n = idx / CC;
    int c = idx - n * CC;

    float acc = 0.f;
#pragma unroll
    for (int k = 0; k < KTOP; ++k) {
        int   e    = tok_e[2 * n + k];
        float p    = tok_p[2 * n + k];
        int   slot = tok_slot[2 * n + k];
        acc += p * bf16_to_f32(eobuf[((size_t)e * NTOK + slot) * CC + c]);
    }
    y[idx] = acc;
}

// ---------------------------------------------------------------------------
// Launcher
// ---------------------------------------------------------------------------
extern "C" void kernel_launch(void* const* d_in, const int* in_sizes, int n_in,
                              void* d_out, int out_size, void* d_ws, size_t ws_size,
                              hipStream_t stream) {
    const float* x  = (const float*)d_in[0];   // [B,T,C]
    const float* Wg = (const float*)d_in[1];   // [C,E]
    const float* W1 = (const float*)d_in[2];   // [E,C,H]
    const float* W2 = (const float*)d_in[3];   // [E,H,C]
    float* y = (float*)d_out;                  // [B,T,C]

    char* ws = (char*)d_ws;
    size_t off = 0;
    auto alloc = [&](size_t bytes) {
        char* p = ws + off;
        off += (bytes + 255) & ~(size_t)255;
        return p;
    };
    unsigned short* xb    = (unsigned short*)alloc((size_t)NTOK * CC * 2);       //   8 MB
    unsigned short* W1b   = (unsigned short*)alloc((size_t)EE * CC * HH * 2);    //  32 MB
    unsigned short* W2b   = (unsigned short*)alloc((size_t)EE * HH * CC * 2);    //  32 MB
    unsigned short* hbuf  = (unsigned short*)alloc((size_t)EE * NTOK * HH * 2);  // 128 MB
    unsigned short* eobuf = (unsigned short*)alloc((size_t)EE * NTOK * CC * 2);  //  64 MB
    int*            counts   = (int*)alloc(EE * sizeof(int));
    int*            tok_idx  = (int*)alloc((size_t)EE * NTOK * sizeof(int));
    int*            tok_e    = (int*)alloc((size_t)NTOK * 2 * sizeof(int));
    float*          tok_p    = (float*)alloc((size_t)NTOK * 2 * sizeof(float));
    int*            tok_slot = (int*)alloc((size_t)NTOK * 2 * sizeof(int));
    (void)ws_size; (void)in_sizes; (void)n_in; (void)out_size;

    // 1) convert to bf16
    {
        int n4 = (NTOK * CC) / 4;
        cvt_f32_bf16_kernel<<<(n4 + 255) / 256, 256, 0, stream>>>(x, xb, n4);
    }
    {
        int n4 = (EE * CC * HH) / 4;
        cvt_f32_bf16_kernel<<<(n4 + 255) / 256, 256, 0, stream>>>(W1, W1b, n4);
    }
    {
        int n4 = (EE * HH * CC) / 4;
        cvt_f32_bf16_kernel<<<(n4 + 255) / 256, 256, 0, stream>>>(W2, W2b, n4);
    }

    // 2) reset per-expert counters
    zero_counts_kernel<<<1, 32, 0, stream>>>(counts);

    // 3) route
    router_kernel<<<NTOK / 128, 128, 0, stream>>>(x, Wg, counts, tok_idx,
                                                  tok_e, tok_p, tok_slot);

    // 4) expert FFN layer 1 (+silu): h = silu(x_g @ W1[e])
    {
        dim3 grid(HH / 256, NTOK / 32, EE);
        ffn_kernel<CC, HH, true, true><<<grid, 256, 0, stream>>>(
            xb, W1b, counts, tok_idx, hbuf);
    }

    // 5) expert FFN layer 2: eo = h @ W2[e]
    {
        dim3 grid(CC / 256, NTOK / 32, EE);
        ffn_kernel<HH, CC, false, false><<<grid, 256, 0, stream>>>(
            hbuf, W2b, counts, tok_idx, eobuf);
    }

    // 6) weighted combine into output
    combine_kernel<<<(NTOK * CC) / 256, 256, 0, stream>>>(eobuf, tok_e, tok_p,
                                                          tok_slot, y);
}
